// LSTMN_4904852652062
// MI455X (gfx1250) — matrix operand order repaired and checked by reading, not compiled, not run on hardware
//
#include <hip/hip_runtime.h>

// ---------------- problem constants ----------------
constexpr int H     = 1024;       // hidden = input size
constexpr int G4    = 4096;       // 4*H gates
constexpr int TT    = 512;        // T
constexpr int TAPE_N = 30;
constexpr int NSTEP = TT - TAPE_N;  // 482 sequential steps
constexpr int NBLK  = 64;         // persistent grid
constexpr int NTHR  = 256;        // 8 waves/block -> 512 waves total

// ---------------- workspace layout (bytes, all 256B aligned) ----------------
constexpr size_t O_WIH   = 0;                              // 2*4096*1024 bf16
constexpr size_t O_WHH   = O_WIH   + (size_t)2*G4*H*2;
constexpr size_t O_WX    = O_WHH   + (size_t)2*G4*H*2;     // 2*1024*1024 bf16
constexpr size_t O_XSB   = O_WX    + (size_t)2*H*H*2;      // 512*1024 bf16
constexpr size_t O_XIH0  = O_XSB   + (size_t)TT*H*2;       // 512*4096 f32
constexpr size_t O_XATTN = O_XIH0  + (size_t)TT*G4*4;      // 2*512*1024 f32
constexpr size_t O_BIAS  = O_XATTN + (size_t)2*TT*H*4;     // 2*4096 f32
constexpr size_t O_DWH   = O_BIAS  + (size_t)2*G4*4;       // 2*1024 f32
constexpr size_t O_DWHT  = O_DWH   + (size_t)2*H*4;
constexpr size_t O_TAPEH = O_DWHT  + (size_t)2*H*4;        // 30*2*1024 f32 ring
constexpr size_t O_TAPEC = O_TAPEH + (size_t)TAPE_N*2*H*4;
constexpr size_t O_PREVH = O_TAPEC + (size_t)TAPE_N*2*H*4; // 2*1024 f32 (= new_h)
constexpr size_t O_NEWC  = O_PREVH + (size_t)2*H*4;
constexpr size_t O_NEWHBF= O_NEWC  + (size_t)2*H*4;        // 2*1024 bf16
constexpr size_t O_H0BF  = O_NEWHBF+ (size_t)2*H*2;        // 1024 bf16
constexpr size_t O_G0P   = O_H0BF  + (size_t)H*2;          // 4096 f32
constexpr size_t O_G1P   = O_G0P   + (size_t)G4*4;
constexpr size_t O_G1P2  = O_G1P   + (size_t)G4*4;
constexpr size_t O_SC    = O_G1P2  + (size_t)G4*4;         // 60 f32 (pad 256)
constexpr size_t O_P1H0  = O_SC    + 256;                  // 32*1024 bf16 (padded)
constexpr size_t O_G1PAD = O_P1H0  + (size_t)32*H*2;       // 32*4096 f32
constexpr size_t O_CNT   = O_G1PAD + (size_t)32*G4*4;      // barrier counter

typedef __attribute__((ext_vector_type(16))) __bf16 v16bf;
typedef __attribute__((ext_vector_type(8)))  __bf16 v8bf;
typedef __attribute__((ext_vector_type(8)))  float  v8f;

__device__ __forceinline__ float sigm(float x) { return 1.f / (1.f + __expf(-x)); }

// ---------------- helpers ----------------
__global__ void k_cvt_bf16(const float* __restrict__ in, __bf16* __restrict__ out, int n) {
  int i = blockIdx.x * blockDim.x + threadIdx.x;
  if (i < n) out[i] = (__bf16)in[i];
}

__global__ void k_prep(const float* __restrict__ b_ih, const float* __restrict__ b_hh,
                       const float* __restrict__ wh,   const float* __restrict__ wht,
                       float* __restrict__ bias, float* __restrict__ dwh, float* __restrict__ dwht) {
  int i = blockIdx.x * blockDim.x + threadIdx.x;
  if (i < 2 * G4) bias[i] = b_ih[i] + b_hh[i];
  if (i < 2 * H) {
    int l = i >> 10, h = i & (H - 1);
    dwh[i]  = wh [(size_t)l * H * H + (size_t)h * H + h];   // diagonal(attn_wh)
    dwht[i] = wht[(size_t)l * H * H + (size_t)h * H + h];   // diagonal(attn_wht)
  }
}

// ---------------- bf16 WMMA GEMM: C[n*M+m] = sum_k W[m,k]*X[n,k] ----------------
// One wave computes one 16x16 tile. A = W tile (lanes 0-15 K 0..15,
// lanes 16-31 K 16..31). B = X tile (lane n holds 16 contiguous K).
__global__ void k_gemm_bf16(const __bf16* __restrict__ W, const __bf16* __restrict__ X,
                            float* __restrict__ C, int M, int K) {
  const int lane = threadIdx.x;
  const int m0 = blockIdx.x * 16, n0 = blockIdx.y * 16;
  const int hi = lane >> 4, lo = lane & 15;
  const __bf16* wp = W + (size_t)(m0 + lo) * K + hi * 16;
  const __bf16* xp = X + (size_t)(n0 + lo) * K + hi * 16;
  v8f acc = {};
  for (int k = 0; k < K; k += 32) {
    v16bf a = *(const v16bf*)(wp + k);
    v16bf b = *(const v16bf*)(xp + k);
    acc = __builtin_amdgcn_wmma_f32_16x16x32_bf16(false, a, false, b, (short)0, acc, false, false);
  }
  // D layout: lanes 0-15 N=lane M=v..v+7 ; lanes 16-31 N=lane-16 M=v+8..
  float* cp = C + (size_t)(n0 + lo) * M + m0 + hi * 8;
  *(v8f*)cp = acc;
}

// ---------------- WMMA GEMV tile: out[m0..m0+15] += nothing prior; B col0 = x ------
__device__ __forceinline__ void gemv_tile(const __bf16* __restrict__ W, const __bf16* __restrict__ x,
                                          float* __restrict__ out, int m0, int lane) {
  const int hi = lane >> 4, lo = lane & 15;
  const __bf16* wp = W + (size_t)(m0 + lo) * H + hi * 16;
  v8f acc = {};
#pragma unroll 4
  for (int k = 0; k < H; k += 32) {
    v16bf a = *(const v16bf*)(wp + k);
    v16bf b = {};                       // only column 0 of B carries the vector
    if (lane == 0)       b = *(const v16bf*)(x + k);        // K 0..15, col 0
    else if (lane == 16) b = *(const v16bf*)(x + k + 16);   // K 16..31, col 0
    acc = __builtin_amdgcn_wmma_f32_16x16x32_bf16(false, a, false, b, (short)0, acc, false, false);
  }
  if (lane == 0)       *(v8f*)(out + m0)     = acc;   // rows m0..m0+7
  else if (lane == 16) *(v8f*)(out + m0 + 8) = acc;   // rows m0+8..m0+15
}

// ---------------- phase 1 (first 30 steps from zero state) ----------------
__global__ void k_phase1a(const float* __restrict__ xih0, const float* __restrict__ bias,
                          float* __restrict__ tapeh, float* __restrict__ tapec,
                          __bf16* __restrict__ p1h0) {
  int tid = blockIdx.x * blockDim.x + threadIdx.x;
  if (tid >= TAPE_N * H) return;
  int t = tid >> 10, h = tid & (H - 1);
  const float* g = xih0 + (size_t)t * G4;
  float gi = g[h] + bias[h], gf = g[H + h] + bias[H + h];
  float gg = g[2 * H + h] + bias[2 * H + h], go = g[3 * H + h] + bias[3 * H + h];
  (void)gf;                                   // c_prev == 0
  float c  = sigm(gi) * tanhf(gg);
  float hh = sigm(go) * tanhf(c);
  tapeh[t * 2 * H + h] = hh;  tapec[t * 2 * H + h] = c;
  p1h0[t * H + h] = (__bf16)hh;
}

__global__ void k_phase1b(const float* __restrict__ g1pad, const float* __restrict__ bias,
                          float* __restrict__ tapeh, float* __restrict__ tapec) {
  int tid = blockIdx.x * blockDim.x + threadIdx.x;
  if (tid >= TAPE_N * H) return;
  int t = tid >> 10, h = tid & (H - 1);
  const float* g = g1pad + (size_t)t * G4;
  const float* b1 = bias + G4;
  float gi = g[h] + b1[h];
  float gg = g[2 * H + h] + b1[2 * H + h], go = g[3 * H + h] + b1[3 * H + h];
  float c  = sigm(gi) * tanhf(gg);
  float hh = sigm(go) * tanhf(c);
  tapeh[t * 2 * H + H + h] = hh;  tapec[t * 2 * H + H + h] = c;
}

// ---------------- device-wide barrier (persistent kernel) ----------------
__device__ __forceinline__ void gbar(unsigned* cnt, unsigned& epoch) {
  __syncthreads();
  epoch += 1;
  if (threadIdx.x == 0) {
    __threadfence();
    __hip_atomic_fetch_add(cnt, 1u, __ATOMIC_RELEASE, __HIP_MEMORY_SCOPE_AGENT);
    const unsigned target = (unsigned)gridDim.x * epoch;
    while (__hip_atomic_load(cnt, __ATOMIC_ACQUIRE, __HIP_MEMORY_SCOPE_AGENT) < target)
      __builtin_amdgcn_s_sleep(1);
  }
  __syncthreads();
  __threadfence();
}

// ---------------- the 482-step sequential scan, one persistent launch ----------------
__global__ void __launch_bounds__(NTHR)
k_seq(char* __restrict__ ws, const float* __restrict__ attn_v, float* __restrict__ out) {
  float*  xih0   = (float*) (ws + O_XIH0);
  float*  xattn  = (float*) (ws + O_XATTN);
  float*  bias   = (float*) (ws + O_BIAS);
  float*  dwh    = (float*) (ws + O_DWH);
  float*  dwht   = (float*) (ws + O_DWHT);
  float*  tapeh  = (float*) (ws + O_TAPEH);
  float*  tapec  = (float*) (ws + O_TAPEC);
  float*  prevh  = (float*) (ws + O_PREVH);
  float*  newc   = (float*) (ws + O_NEWC);
  __bf16* newhbf = (__bf16*)(ws + O_NEWHBF);
  __bf16* h0bf   = (__bf16*)(ws + O_H0BF);
  float*  g0p    = (float*) (ws + O_G0P);
  float*  g1p    = (float*) (ws + O_G1P);
  float*  g1p2   = (float*) (ws + O_G1P2);
  float*  sc     = (float*) (ws + O_SC);
  const __bf16* whh  = (const __bf16*)(ws + O_WHH);
  const __bf16* wih1 = (const __bf16*)(ws + O_WIH) + (size_t)G4 * H;   // layer-1 W_ih
  unsigned* cnt = (unsigned*)(ws + O_CNT);

  const int tid  = blockIdx.x * blockDim.x + threadIdx.x;
  const int lane = threadIdx.x & 31;
  const int wave = tid >> 5;                 // 0..511
  unsigned epoch = 0;
  __shared__ float red[NTHR];

  for (int s = 0; s < NSTEP; ++s) {
    const int xt = TAPE_N + s;
    // ---- a1: attention scores[j,l] = sum_h tanh(dwh*h_tape + x_attn + dwht*prev_h)*v
    if (blockIdx.x < 2 * TAPE_N) {
      const int l = blockIdx.x / TAPE_N, j = blockIdx.x % TAPE_N;
      const int slot = (s + j) % TAPE_N;
      const float* th = tapeh + slot * 2 * H + l * H;
      const float* xa = xattn + ((size_t)l * TT + xt) * H;
      const float* ph = prevh + l * H;
      const float* dh = dwh + l * H;
      const float* dt = dwht + l * H;
      const float* av = attn_v + l * H;
      float p = 0.f;
      for (int h = threadIdx.x; h < H; h += NTHR)
        p += tanhf(dh[h] * th[h] + xa[h] + dt[h] * ph[h]) * av[h];
      red[threadIdx.x] = p; __syncthreads();
      for (int off = NTHR / 2; off > 0; off >>= 1) {
        if (threadIdx.x < off) red[threadIdx.x] += red[threadIdx.x + off];
        __syncthreads();
      }
      if (threadIdx.x == 0) sc[l * TAPE_N + j] = red[0];
    }
    gbar(cnt, epoch);
    // ---- a2: softmax over tape + weighted sums -> new_h (=prev_h for next step), new_c
    if (tid < 2 * H) {
      const int l = tid >> 10, h = tid & (H - 1);
      const float* scl = sc + l * TAPE_N;
      float mx = scl[0];
#pragma unroll
      for (int j = 1; j < TAPE_N; ++j) mx = fmaxf(mx, scl[j]);
      float al[TAPE_N]; float se = 0.f;
#pragma unroll
      for (int j = 0; j < TAPE_N; ++j) { al[j] = __expf(scl[j] - mx); se += al[j]; }
      const float inv = 1.f / se;
      float nh = 0.f, ncv = 0.f;
      int slot = s % TAPE_N;
#pragma unroll
      for (int j = 0; j < TAPE_N; ++j) {
        const float w = al[j] * inv;
        nh  += w * tapeh[slot * 2 * H + l * H + h];
        ncv += w * tapec[slot * 2 * H + l * H + h];
        slot = (slot + 1 == TAPE_N) ? 0 : slot + 1;
      }
      prevh[l * H + h]  = nh;
      newc [l * H + h]  = ncv;
      newhbf[l * H + h] = (__bf16)nh;
    }
    gbar(cnt, epoch);
    // ---- b: g0p = W_hh0 @ new_h0 ; g1p = W_hh1 @ new_h1   (512 tiles = 512 waves)
    {
      const int l = wave >> 8, m0 = (wave & 255) * 16;
      gemv_tile(whh + (size_t)l * G4 * H, newhbf + l * H, (l == 0 ? g0p : g1p), m0, lane);
    }
    gbar(cnt, epoch);
    // ---- c: layer-0 elementwise LSTM (gates = Xih0[t] + bias0 + g0p, c_prev = new_c0)
    if (tid < H) {
      const int h = tid;
      const float* gx = xih0 + (size_t)xt * G4;
      float gi = gx[h]         + bias[h]         + g0p[h];
      float gf = gx[H + h]     + bias[H + h]     + g0p[H + h];
      float gg = gx[2 * H + h] + bias[2 * H + h] + g0p[2 * H + h];
      float go = gx[3 * H + h] + bias[3 * H + h] + g0p[3 * H + h];
      float c  = sigm(gf) * newc[h] + sigm(gi) * tanhf(gg);
      float hh = sigm(go) * tanhf(c);
      const int slot = s % TAPE_N;
      tapeh[slot * 2 * H + h] = hh;
      tapec[slot * 2 * H + h] = c;
      h0bf[h] = (__bf16)hh;
    }
    gbar(cnt, epoch);
    // ---- d: g1p2 = W_ih1 @ h0   (256 tiles on waves 0..255)
    if (wave < 256) gemv_tile(wih1, h0bf, g1p2, wave * 16, lane);
    gbar(cnt, epoch);
    // ---- e: layer-1 elementwise LSTM + emit output row
    if (tid < H) {
      const int h = tid;
      const float* b1 = bias + G4;
      float gi = g1p[h]         + g1p2[h]         + b1[h];
      float gf = g1p[H + h]     + g1p2[H + h]     + b1[H + h];
      float gg = g1p[2 * H + h] + g1p2[2 * H + h] + b1[2 * H + h];
      float go = g1p[3 * H + h] + g1p2[3 * H + h] + b1[3 * H + h];
      float c  = sigm(gf) * newc[H + h] + sigm(gi) * tanhf(gg);
      float hh = sigm(go) * tanhf(c);
      const int slot = s % TAPE_N;
      tapeh[slot * 2 * H + H + h] = hh;
      tapec[slot * 2 * H + H + h] = c;
      out[(size_t)s * H + h] = hh;
    }
    gbar(cnt, epoch);
  }
}

// ---------------- host ----------------
extern "C" void kernel_launch(void* const* d_in, const int* in_sizes, int n_in,
                              void* d_out, int out_size, void* d_ws, size_t ws_size,
                              hipStream_t stream) {
  (void)in_sizes; (void)n_in; (void)out_size; (void)ws_size;
  const float* xs   = (const float*)d_in[0];
  const float* w_ih = (const float*)d_in[1];
  const float* w_hh = (const float*)d_in[2];
  const float* b_ih = (const float*)d_in[3];
  const float* b_hh = (const float*)d_in[4];
  const float* awh  = (const float*)d_in[5];
  const float* awx  = (const float*)d_in[6];
  const float* awht = (const float*)d_in[7];
  const float* av   = (const float*)d_in[8];
  float* out = (float*)d_out;
  char*  ws  = (char*)d_ws;

  // deterministic init (graph-capturable)
  hipMemsetAsync(ws + O_CNT,   0, 64, stream);
  hipMemsetAsync(ws + O_PREVH, 0, (size_t)2 * H * sizeof(float), stream);   // prev_h = 0
  hipMemsetAsync(ws + O_P1H0,  0, (size_t)32 * H * sizeof(__bf16), stream); // N-pad rows
  hipMemsetAsync(d_out,        0, (size_t)TT * H * sizeof(float), stream);  // rows 482..511 stay 0

  // bf16 copies of streamed weights / inputs (halves per-step L2 traffic)
  k_cvt_bf16<<<(2 * G4 * H + 255) / 256, 256, 0, stream>>>(w_ih, (__bf16*)(ws + O_WIH), 2 * G4 * H);
  k_cvt_bf16<<<(2 * G4 * H + 255) / 256, 256, 0, stream>>>(w_hh, (__bf16*)(ws + O_WHH), 2 * G4 * H);
  k_cvt_bf16<<<(2 * H * H + 255) / 256, 256, 0, stream>>>(awx,  (__bf16*)(ws + O_WX),  2 * H * H);
  k_cvt_bf16<<<(TT * H + 255) / 256, 256, 0, stream>>>(xs,      (__bf16*)(ws + O_XSB), TT * H);
  k_prep<<<(2 * G4 + 255) / 256, 256, 0, stream>>>(b_ih, b_hh, awh, awht,
      (float*)(ws + O_BIAS), (float*)(ws + O_DWH), (float*)(ws + O_DWHT));

  const __bf16* wihb = (const __bf16*)(ws + O_WIH);
  const __bf16* wxb  = (const __bf16*)(ws + O_WX);
  const __bf16* xsb  = (const __bf16*)(ws + O_XSB);

  // Hoisted GEMMs (full WMMA tiles, N = 512 timesteps)
  k_gemm_bf16<<<dim3(G4 / 16, TT / 16), 32, 0, stream>>>(wihb, xsb, (float*)(ws + O_XIH0), G4, H);
  k_gemm_bf16<<<dim3(H / 16, TT / 16), 32, 0, stream>>>(wxb, xsb, (float*)(ws + O_XATTN), H, H);
  k_gemm_bf16<<<dim3(H / 16, TT / 16), 32, 0, stream>>>(wxb + (size_t)H * H, xsb,
      (float*)(ws + O_XATTN) + (size_t)TT * H, H, H);

  // Phase 1: first 30 steps from zero state -> initial tape
  k_phase1a<<<(TAPE_N * H + 255) / 256, 256, 0, stream>>>((const float*)(ws + O_XIH0),
      (const float*)(ws + O_BIAS), (float*)(ws + O_TAPEH), (float*)(ws + O_TAPEC),
      (__bf16*)(ws + O_P1H0));
  k_gemm_bf16<<<dim3(G4 / 16, 2), 32, 0, stream>>>(wihb + (size_t)G4 * H,
      (const __bf16*)(ws + O_P1H0), (float*)(ws + O_G1PAD), G4, H);
  k_phase1b<<<(TAPE_N * H + 255) / 256, 256, 0, stream>>>((const float*)(ws + O_G1PAD),
      (const float*)(ws + O_BIAS), (float*)(ws + O_TAPEH), (float*)(ws + O_TAPEC));

  // Sequential 482-step scan: one persistent launch, device-wide atomic barriers
  k_seq<<<NBLK, NTHR, 0, stream>>>(ws, av, out);
}
